// GATLayer_10797547782754
// MI455X (gfx1250) — matrix operand order repaired and checked by reading, not compiled, not run on hardware
//
#include <hip/hip_runtime.h>
#include <hip/hip_bf16.h>

typedef __attribute__((ext_vector_type(16))) _Float16 v16h;
typedef __attribute__((ext_vector_type(8)))  float    v8f;
typedef __attribute__((ext_vector_type(4)))  unsigned int v4u;
typedef __attribute__((ext_vector_type(8)))  int      v8i;
typedef __attribute__((ext_vector_type(4)))  int      v4i;

#define N_NODES 8192
#define F_IN    128
#define F_OUT   64
#define JSPLIT  8
#define GAT_ALPHA 0.2f
#define NEG_INF (-9.0e15f)
#define LDS_ROW 36   // 32 data floats + 4 pad floats (TDM pad) -> conflict-free ds_load_b128

#define UNPACK4F(dst, off, v) { dst[(off)+0]=(v).x; dst[(off)+1]=(v).y; dst[(off)+2]=(v).z; dst[(off)+3]=(v).w; }
#define UNPACK4I(dst, off, v) { dst[(off)+0]=(v).x; dst[(off)+1]=(v).y; dst[(off)+2]=(v).z; dst[(off)+3]=(v).w; }

// ---------------------------------------------------------------------------
// TDM: load a 16-row x 32-col tile of 4-byte elements (row stride 8192 elems)
// from global into LDS, padding each 32-DWORD row with 4 DWORDs (LDS_ROW=36).
// Descriptor bitfields per CDNA5 ISA 8.3/8.4; 6-arg (clang-23) builtin form.
// ---------------------------------------------------------------------------
__device__ __forceinline__ void tdm_load_tile(unsigned lds_off, const void* gptr) {
  const unsigned long long ga = (unsigned long long)(uintptr_t)gptr;
  v4u g0;
  g0[0] = 1u;                                       // count=1, user mode, no gather
  g0[1] = lds_off;                                  // lds_addr (bytes, wave-relative)
  g0[2] = (unsigned)(ga & 0xFFFFFFFFu);             // global_addr[31:0]
  g0[3] = (unsigned)((ga >> 32) & 0x01FFFFFFu)      // global_addr[56:32]
        | (2u << 30);                               // type = 2 (image)
  v8i g1;
  g1[0] = (int)((2u << 16)                          // data_size = 2 -> 4 bytes
        | (1u << 20)                                // pad_enable
        | (4u << 22)                                // pad_interval: 1<<4 * 8B = 32 DWORDs
        | (3u << 25));                              // pad_amount: 3 -> 4 DWORDs
  g1[1] = (int)((N_NODES & 0xFFFFu) << 16);         // tensor_dim0[15:0] (=8192)
  g1[2] = (int)(((N_NODES >> 16) & 0xFFFFu)         // tensor_dim0[31:16]
        | ((N_NODES & 0xFFFFu) << 16));             // tensor_dim1[15:0]
  g1[3] = (int)(((N_NODES >> 16) & 0xFFFFu)         // tensor_dim1[31:16]
        | (32u << 16));                             // tile_dim0 = 32
  g1[4] = (int)16;                                  // tile_dim1 = 16, tile_dim2 = 0
  g1[5] = (int)N_NODES;                             // tensor_dim0_stride[31:0]
  g1[6] = 0;                                        // stride hi + dim1_stride lo
  g1[7] = 0;
  const v4i z4 = {0, 0, 0, 0};                      // groups 2/3 unused (2D tile)
  const v8i z8 = {0, 0, 0, 0, 0, 0, 0, 0};          // extra group (clang-23 form)
  __builtin_amdgcn_tensor_load_to_lds(g0, g1, z4, z4, z8, 0);
}

// ---------------------------------------------------------------------------
// Kernel 1: Wh = h @ W  (store transposed f16 for WMMA-B), s1 = Wh@a1, s2 = Wh@a2
// ---------------------------------------------------------------------------
__global__ void __launch_bounds__(64)
gat_wh_kernel(const float* __restrict__ h, const float* __restrict__ W,
              const float* __restrict__ a,
              _Float16* __restrict__ WhT,
              float* __restrict__ s1, float* __restrict__ s2) {
  __shared__ float hrow[F_IN];
  __shared__ float red1[F_OUT];
  __shared__ float red2[F_OUT];
  const int i = blockIdx.x;
  const int t = threadIdx.x;           // 0..63
  hrow[t]      = h[(size_t)i * F_IN + t];
  hrow[t + 64] = h[(size_t)i * F_IN + t + 64];
  __syncthreads();
  float acc = 0.f;
  #pragma unroll 8
  for (int k = 0; k < F_IN; ++k)
    acc = fmaf(hrow[k], W[k * F_OUT + t], acc);
  WhT[(size_t)t * N_NODES + i] = (_Float16)acc;
  red1[t] = acc * a[t];
  red2[t] = acc * a[F_OUT + t];
  __syncthreads();
  if (t == 0) {
    float x1 = 0.f, x2 = 0.f;
    #pragma unroll
    for (int k = 0; k < F_OUT; ++k) { x1 += red1[k]; x2 += red2[k]; }
    s1[i] = x1;
    s2[i] = x2;
  }
}

// ---------------------------------------------------------------------------
// Kernel 2: fused masked-softmax attention, flash style, split over columns.
// One wave per (16-row tile, 1024-col segment). edge/adj tiles staged through
// LDS by the Tensor Data Mover (double buffered, TENSORcnt-synchronized).
// ---------------------------------------------------------------------------
__global__ void __launch_bounds__(32)
gat_attn_kernel(const float* __restrict__ edge, const int* __restrict__ adj,
                const float* __restrict__ a,
                const float* __restrict__ s1, const float* __restrict__ s2,
                const _Float16* __restrict__ WhT,
                float* __restrict__ Cpart, float* __restrict__ Mpart,
                float* __restrict__ Spart) {
  __shared__ __align__(16) float ebuf[2][16 * LDS_ROW];
  __shared__ __align__(16) int   abuf[2][16 * LDS_ROW];

  const int tile = blockIdx.x / JSPLIT;
  const int js   = blockIdx.x % JSPLIT;
  const int l    = threadIdx.x;
  const int g    = l >> 4;            // lane group (0/1)
  const int m    = l & 15;            // A row (and B column) index
  const int r0   = tile * 16;
  const int row  = r0 + m;
  const int jseg = N_NODES / JSPLIT;  // 1024 columns per segment
  const int j0   = js * jseg;
  const int NIT  = jseg / 32;         // 32 chunks

  const float ae  = a[2 * F_OUT];
  const float s1r = s1[row];

  const float* __restrict__ ebase = edge + (size_t)r0 * N_NODES;
  const int*   __restrict__ abase = adj  + (size_t)r0 * N_NODES;

  float rmax = NEG_INF;   // running row max (per lane-row, duplicated l / l+16)
  float rsum = 0.f;       // running row sum of exp
  v8f c0 = {}, c1 = {}, c2 = {}, c3 = {};

  // prologue: kick off first tile pair
  {
    const unsigned eoff = (unsigned)(uintptr_t)(void*)&ebuf[0][0];
    const unsigned aoff = (unsigned)(uintptr_t)(void*)&abuf[0][0];
    tdm_load_tile(eoff, ebase + j0);
    tdm_load_tile(aoff, abase + j0);
  }

  for (int it = 0; it < NIT; ++it) {
    const int jb = j0 + it * 32;
    if (it + 1 < NIT) {
      const int nb = (it + 1) & 1;
      const unsigned eoff = (unsigned)(uintptr_t)(void*)&ebuf[nb][0];
      const unsigned aoff = (unsigned)(uintptr_t)(void*)&abuf[nb][0];
      tdm_load_tile(eoff, ebase + jb + 32);
      tdm_load_tile(aoff, abase + jb + 32);
      __builtin_amdgcn_s_wait_tensorcnt(2);   // current pair complete
    } else {
      __builtin_amdgcn_s_wait_tensorcnt(0);
    }
    const int b = it & 1;

    // --- read this lane's 16 columns from LDS in WMMA-A order:
    //     K(t) = t + 8g (t<8) ; K(t) = t + 8 + 8g (t>=8)
    const float* ep = &ebuf[b][m * LDS_ROW + 8 * g];
    const int*   ap = &abuf[b][m * LDS_ROW + 8 * g];
    float ev[16]; int mk[16]; float sv[16];
    { float4 v;
      v = *(const float4*)(ep);      UNPACK4F(ev, 0,  v);
      v = *(const float4*)(ep + 4);  UNPACK4F(ev, 4,  v);
      v = *(const float4*)(ep + 16); UNPACK4F(ev, 8,  v);
      v = *(const float4*)(ep + 20); UNPACK4F(ev, 12, v);
    }
    { int4 v;
      v = *(const int4*)(ap);        UNPACK4I(mk, 0,  v);
      v = *(const int4*)(ap + 4);    UNPACK4I(mk, 4,  v);
      v = *(const int4*)(ap + 16);   UNPACK4I(mk, 8,  v);
      v = *(const int4*)(ap + 20);   UNPACK4I(mk, 12, v);
    }
    { const int o0 = jb + 8 * g;
      const int o1 = jb + 16 + 8 * g;
      float4 v;
      v = *(const float4*)(s2 + o0);     UNPACK4F(sv, 0,  v);
      v = *(const float4*)(s2 + o0 + 4); UNPACK4F(sv, 4,  v);
      v = *(const float4*)(s2 + o1);     UNPACK4F(sv, 8,  v);
      v = *(const float4*)(s2 + o1 + 4); UNPACK4F(sv, 12, v);
    }

    // --- scores: e = s1[i] + s2[j] + a_e*edge; leaky-relu; mask
    float val[16];
    float cmax = NEG_INF;
    #pragma unroll
    for (int t = 0; t < 16; ++t) {
      float e = fmaf(ae, ev[t], s1r + sv[t]);
      e = (e > 0.f) ? e : GAT_ALPHA * e;
      val[t] = (mk[t] > 0) ? e : NEG_INF;
      cmax = fmaxf(cmax, val[t]);
    }
    cmax = fmaxf(cmax, __shfl_xor(cmax, 16, 32));
    const float nmax  = fmaxf(rmax, cmax);
    const float scale = __expf(rmax - nmax);   // ->0 if rmax was NEG_INF

    // --- probabilities into WMMA A operand (f16)
    v16h A;
    float psum = 0.f;
    #pragma unroll
    for (int t = 0; t < 16; ++t) {
      const float p = (mk[t] > 0) ? __expf(val[t] - nmax) : 0.f;
      psum += p;
      A[t] = (_Float16)p;
    }
    psum += __shfl_xor(psum, 16, 32);
    rsum = rsum * scale + psum;
    rmax = nmax;

    // --- rescale accumulators only when some row max actually moved.
    //     C VGPR r holds row (8g + r); fetch that row's scale from lane 8g+r.
    if (__any(scale < 1.0f)) {
      float sc[8];
      #pragma unroll
      for (int r = 0; r < 8; ++r) sc[r] = __shfl(scale, 8 * g + r, 32);
      #pragma unroll
      for (int r = 0; r < 8; ++r) {
        c0[r] *= sc[r]; c1[r] *= sc[r]; c2[r] *= sc[r]; c3[r] *= sc[r];
      }
    }

    // --- B operand: WhT[n][jb+8g .. +7] and WhT[n][jb+16+8g .. +7] per lane (L2-resident)
    const _Float16* bbase = WhT + jb + 8 * g;
    {
      v16h B;
      ((float4*)&B)[0] = *(const float4*)(bbase + (size_t)(0 * 16 + m) * N_NODES);
      ((float4*)&B)[1] = *(const float4*)(bbase + (size_t)(0 * 16 + m) * N_NODES + 16);
      c0 = __builtin_amdgcn_wmma_f32_16x16x32_f16(false, A, false, B, (short)0, c0, false, false);
    }
    {
      v16h B;
      ((float4*)&B)[0] = *(const float4*)(bbase + (size_t)(1 * 16 + m) * N_NODES);
      ((float4*)&B)[1] = *(const float4*)(bbase + (size_t)(1 * 16 + m) * N_NODES + 16);
      c1 = __builtin_amdgcn_wmma_f32_16x16x32_f16(false, A, false, B, (short)0, c1, false, false);
    }
    {
      v16h B;
      ((float4*)&B)[0] = *(const float4*)(bbase + (size_t)(2 * 16 + m) * N_NODES);
      ((float4*)&B)[1] = *(const float4*)(bbase + (size_t)(2 * 16 + m) * N_NODES + 16);
      c2 = __builtin_amdgcn_wmma_f32_16x16x32_f16(false, A, false, B, (short)0, c2, false, false);
    }
    {
      v16h B;
      ((float4*)&B)[0] = *(const float4*)(bbase + (size_t)(3 * 16 + m) * N_NODES);
      ((float4*)&B)[1] = *(const float4*)(bbase + (size_t)(3 * 16 + m) * N_NODES + 16);
      c3 = __builtin_amdgcn_wmma_f32_16x16x32_f16(false, A, false, B, (short)0, c3, false, false);
    }
  }

  // --- write partial tile: C layout => VGPR r, lane l: row = r + 8g, col = l%16
  float* cp = Cpart + ((size_t)tile * JSPLIT + js) * 16 * F_OUT;
  #pragma unroll
  for (int r = 0; r < 8; ++r) {
    const int rr = r + 8 * g;
    cp[rr * F_OUT + 0 * 16 + m] = c0[r];
    cp[rr * F_OUT + 1 * 16 + m] = c1[r];
    cp[rr * F_OUT + 2 * 16 + m] = c2[r];
    cp[rr * F_OUT + 3 * 16 + m] = c3[r];
  }
  if (g == 0) {
    Mpart[(size_t)row * JSPLIT + js] = rmax;
    Spart[(size_t)row * JSPLIT + js] = rsum;
  }
}

// ---------------------------------------------------------------------------
// Kernel 3: combine column-split partials (flash merge), normalize, ELU.
// ---------------------------------------------------------------------------
__global__ void __launch_bounds__(64)
gat_reduce_kernel(const float* __restrict__ Cpart, const float* __restrict__ Mpart,
                  const float* __restrict__ Spart, float* __restrict__ out) {
  const int i = blockIdx.x;
  const int t = threadIdx.x;     // output feature
  const int tile = i >> 4;
  const int mm   = i & 15;
  float mstar = NEG_INF;
  #pragma unroll
  for (int js = 0; js < JSPLIT; ++js)
    mstar = fmaxf(mstar, Mpart[(size_t)i * JSPLIT + js]);
  float acc = 0.f, wsum = 0.f;
  #pragma unroll
  for (int js = 0; js < JSPLIT; ++js) {
    const float w = __expf(Mpart[(size_t)i * JSPLIT + js] - mstar);
    wsum += w * Spart[(size_t)i * JSPLIT + js];
    acc  += w * Cpart[(((size_t)tile * JSPLIT + js) * 16 + mm) * F_OUT + t];
  }
  const float x = acc / fmaxf(wsum, 1e-30f);
  out[(size_t)i * F_OUT + t] = (x > 0.f) ? x : (__expf(x) - 1.f);
}

// ---------------------------------------------------------------------------
extern "C" void kernel_launch(void* const* d_in, const int* in_sizes, int n_in,
                              void* d_out, int out_size, void* d_ws, size_t ws_size,
                              hipStream_t stream) {
  (void)in_sizes; (void)n_in; (void)out_size; (void)ws_size;
  const float* h    = (const float*)d_in[0];
  const int*   adj  = (const int*)d_in[1];
  const float* edge = (const float*)d_in[2];
  const float* W    = (const float*)d_in[3];
  const float* a    = (const float*)d_in[4];
  float* out = (float*)d_out;

  char* ws = (char*)d_ws;
  size_t off = 0;
  _Float16* WhT = (_Float16*)(ws + off); off += (size_t)F_OUT * N_NODES * sizeof(_Float16); // 1 MB
  float* s1     = (float*)(ws + off);    off += (size_t)N_NODES * sizeof(float);
  float* s2     = (float*)(ws + off);    off += (size_t)N_NODES * sizeof(float);
  float* Mpart  = (float*)(ws + off);    off += (size_t)N_NODES * JSPLIT * sizeof(float);
  float* Spart  = (float*)(ws + off);    off += (size_t)N_NODES * JSPLIT * sizeof(float);
  float* Cpart  = (float*)(ws + off);    off += (size_t)N_NODES * JSPLIT * F_OUT * sizeof(float); // 16 MB

  gat_wh_kernel<<<N_NODES, 64, 0, stream>>>(h, W, a, WhT, s1, s2);
  gat_attn_kernel<<<(N_NODES / 16) * JSPLIT, 32, 0, stream>>>(
      edge, adj, a, s1, s2, WhT, Cpart, Mpart, Spart);
  gat_reduce_kernel<<<N_NODES, 64, 0, stream>>>(Cpart, Mpart, Spart, out);
}
